// Topograph_32315333935161
// MI455X (gfx1250) — matrix operand order repaired and checked by reading, not compiled, not run on hardware
//
#include <hip/hip_runtime.h>
#include <hip/hip_bf16.h>
#include <stdint.h>

typedef __attribute__((ext_vector_type(16))) __bf16 v16bf;
typedef __attribute__((ext_vector_type(8)))  float  v8f;
typedef unsigned short u16;

__device__ __forceinline__ u16 f32_to_bf16(float f) {
  union { float f; unsigned u; } c; c.f = f;
  unsigned u = c.u;
  unsigned r = u + 0x7fffu + ((u >> 16) & 1u);   // round-to-nearest-even
  return (u16)(r >> 16);
}
__device__ __forceinline__ float bf16_to_f32(u16 h) {
  union { unsigned u; float f; } c; c.u = ((unsigned)h) << 16; return c.f;
}

// ---------------------------------------------------------------------------
// Zero page for OOB halo reads (replaces per-load EXEC predication).
// ---------------------------------------------------------------------------
__global__ void zero_page_kernel(unsigned* zp) { zp[threadIdx.x] = 0u; }

// ---------------------------------------------------------------------------
// Repack tower weights [4][256][256][3][3] f32 -> per-lane WMMA A-layout bf16.
// Layout: [layer][tap][ktile][otile][lane(32)][j(16)]  (16 bf16 = 32B/lane)
// A 16x32 bf16: lanes 0-15 (M=lane): K=j;
//               lanes 16-31:         K=j+16
// ---------------------------------------------------------------------------
__global__ void repack_weights_kernel(const float* __restrict__ w,
                                      u16* __restrict__ out, int total) {
  int idx = blockIdx.x * blockDim.x + threadIdx.x;
  if (idx >= total) return;
  int t = idx;
  const int j    = t & 15;  t >>= 4;
  const int lane = t & 31;  t >>= 5;
  const int ot   = t & 15;  t >>= 4;
  const int kt   = t & 7;   t >>= 3;
  const int tap  = t % 9;   t /= 9;
  const int l    = t;
  const int laneHi = lane >> 4;
  const int m      = lane & 15;
  const int K  = j + 16 * laneHi;
  const int o  = ot * 16 + m;
  const int ci = kt * 32 + K;
  const int kh = tap / 3;
  const int kw = tap - kh * 3;
  out[idx] = f32_to_bf16(w[((((size_t)l * 256 + o) * 256 + ci) * 3 + kh) * 3 + kw]);
}

// ---------------------------------------------------------------------------
// NCHW f32 -> NHWC bf16 (level input conversion, shared by both towers).
// ---------------------------------------------------------------------------
__global__ __launch_bounds__(256)
void nchw_to_nhwc_bf16_kernel(const float* __restrict__ in,
                              u16* __restrict__ out, int HW, int total8) {
  const int idx = blockIdx.x * blockDim.x + threadIdx.x;
  if (idx >= total8) return;
  const int c8  = (idx & 31) * 8;
  const int pos = idx >> 5;
  const int b   = pos / HW;
  const int hw  = pos - b * HW;
  const float* ip = in + ((size_t)b * 256 + c8) * HW + hw;
  u16 v[8];
  #pragma unroll
  for (int j = 0; j < 8; ++j) v[j] = f32_to_bf16(ip[(size_t)j * HW]);
  uint4 o;
  o.x = (unsigned)v[0] | ((unsigned)v[1] << 16);
  o.y = (unsigned)v[2] | ((unsigned)v[3] << 16);
  o.z = (unsigned)v[4] | ((unsigned)v[5] << 16);
  o.w = (unsigned)v[6] | ((unsigned)v[7] << 16);
  *(uint4*)(out + (size_t)pos * 256 + c8) = o;
}

// ---------------------------------------------------------------------------
// Implicit-GEMM conv3x3 (SAME, no bias): NHWC bf16 in, NHWC f32 out.
// Block = 128 threads (4 waves). Block covers 32 positions x 256 out chans.
// Each wave: 2 spatial tiles x 4 otile accumulators (8 WMMAs per K-chunk),
// stage-indexed double buffering of A(x4) and B(x2), zero-page for halo OOB.
// ---------------------------------------------------------------------------
__global__ __launch_bounds__(128)
void conv3x3_wmma_kernel(const u16* __restrict__ xq,
                         const u16* __restrict__ wp,
                         float* __restrict__ y,
                         const u16* __restrict__ zpage,
                         int H, int W, int HW, int npos) {
  const int wave   = threadIdx.x >> 5;
  const int lane   = threadIdx.x & 31;
  const int laneHi = lane >> 4;
  const int ncol   = lane & 15;
  const int base0  = blockIdx.x * 32;

  int bb[2], hh0[2], ww0[2], hwp[2];
  bool valid[2];
  #pragma unroll
  for (int n = 0; n < 2; ++n) {
    const int pos = base0 + n * 16 + ncol;
    valid[n] = pos < npos;
    bb[n]  = pos / HW;
    hwp[n] = pos - bb[n] * HW;
    hh0[n] = hwp[n] / W;
    ww0[n] = hwp[n] - hh0[n] * W;
  }

  v8f acc[2][4] = {};
  const u16* pA0 = wp + wave * 2048 + lane * 16;
  const u16* zp  = zpage + laneHi * 16;

  for (int tap = 0; tap < 9; ++tap) {
    const int dh = tap / 3 - 1;
    const int dw = tap - (tap / 3) * 3 - 1;
    const u16* pB[2];
    #pragma unroll
    for (int n = 0; n < 2; ++n) {
      const int h2 = hh0[n] + dh, w2 = ww0[n] + dw;
      const bool inb = valid[n] && (h2 >= 0) && (h2 < H) && (w2 >= 0) && (w2 < W);
      pB[n] = inb ? xq + ((size_t)bb[n] * HW + (size_t)h2 * W + w2) * 256 + laneHi * 16
                  : zp;
    }
    const u16* pA = pA0 + (size_t)tap * 65536;
    // per-kt strides: A += 8192 elems (16KB), otile += 512; B += 32 elems (64B)

    v16bf Ab[2][4], Bb[2][2];
    #pragma unroll
    for (int a = 0; a < 4; ++a) Ab[0][a] = *(const v16bf*)(pA + a * 512);
    #pragma unroll
    for (int n = 0; n < 2; ++n) Bb[0][n] = *(const v16bf*)(pB[n]);

    #pragma unroll
    for (int kt = 0; kt < 8; ++kt) {
      const int cur = kt & 1, nxt = cur ^ 1;
      if (kt < 7) {
        const u16* pAn = pA + (size_t)(kt + 1) * 8192;
        #pragma unroll
        for (int a = 0; a < 4; ++a) Ab[nxt][a] = *(const v16bf*)(pAn + a * 512);
        #pragma unroll
        for (int n = 0; n < 2; ++n) Bb[nxt][n] = *(const v16bf*)(pB[n] + (kt + 1) * 32);
      }
      #pragma unroll
      for (int n = 0; n < 2; ++n) {
        #pragma unroll
        for (int a = 0; a < 4; ++a) {
          acc[n][a] = __builtin_amdgcn_wmma_f32_16x16x32_bf16(
              false, Ab[cur][a], false, Bb[cur][n], (short)0, acc[n][a], false, false);
        }
      }
    }
  }

  // C/D layout: VGPR r -> M = r + 8*laneHi; NHWC => 8 consecutive channels
  #pragma unroll
  for (int n = 0; n < 2; ++n) {
    if (valid[n]) {
      float* yb = y + ((size_t)bb[n] * HW + hwp[n]) * 256 + laneHi * 8;
      #pragma unroll
      for (int a = 0; a < 4; ++a) {
        *(v8f*)(yb + (wave * 4 + a) * 16) = acc[n][a];
      }
    }
  }
}

// ---------------------------------------------------------------------------
// Fused GroupNorm(32 groups) + affine + ReLU + bf16 NHWC output.
// One block per (batch, group). float4 loads, packed uint2 bf16 stores.
// ---------------------------------------------------------------------------
__global__ __launch_bounds__(256)
void gn_relu_kernel(const float* __restrict__ y,      // NHWC f32
                    const float* __restrict__ gamma,
                    const float* __restrict__ beta,
                    u16* __restrict__ out,            // NHWC bf16
                    int HW) {
  const int b = blockIdx.x >> 5;
  const int g = blockIdx.x & 31;
  const float* base = y   + (size_t)b * HW * 256 + g * 8;
  u16* ob           = out + (size_t)b * HW * 256 + g * 8;
  const int nq = 2 * HW;                      // float4 quads over 8*HW elems
  float s = 0.f, ss = 0.f;
  for (int i = threadIdx.x; i < nq; i += 256) {
    const int posi = i >> 1;
    const int q    = (i & 1) * 4;
    const float4 v = *(const float4*)(base + (size_t)posi * 256 + q);
    s  += v.x + v.y + v.z + v.w;
    ss += v.x * v.x + v.y * v.y + v.z * v.z + v.w * v.w;
  }
  __shared__ float rs[256];
  __shared__ float rq[256];
  rs[threadIdx.x] = s; rq[threadIdx.x] = ss;
  __syncthreads();
  for (int off = 128; off > 0; off >>= 1) {
    if (threadIdx.x < off) {
      rs[threadIdx.x] += rs[threadIdx.x + off];
      rq[threadIdx.x] += rq[threadIdx.x + off];
    }
    __syncthreads();
  }
  const float inv_n = 1.0f / (float)(8 * HW);
  const float mu    = rs[0] * inv_n;
  const float var   = rq[0] * inv_n - mu * mu;
  const float rstd  = rsqrtf(var + 1e-5f);
  float ga[8], be[8];
  #pragma unroll
  for (int c = 0; c < 8; ++c) {
    ga[c] = gamma[g * 8 + c] * rstd;
    be[c] = beta[g * 8 + c] - mu * gamma[g * 8 + c] * rstd;
  }
  for (int i = threadIdx.x; i < nq; i += 256) {
    const int posi = i >> 1;
    const int q    = (i & 1) * 4;
    const float4 v = *(const float4*)(base + (size_t)posi * 256 + q);
    float r0 = v.x * ga[q + 0] + be[q + 0]; r0 = r0 > 0.f ? r0 : 0.f;
    float r1 = v.y * ga[q + 1] + be[q + 1]; r1 = r1 > 0.f ? r1 : 0.f;
    float r2 = v.z * ga[q + 2] + be[q + 2]; r2 = r2 > 0.f ? r2 : 0.f;
    float r3 = v.w * ga[q + 3] + be[q + 3]; r3 = r3 > 0.f ? r3 : 0.f;
    uint2 o;
    o.x = (unsigned)f32_to_bf16(r0) | ((unsigned)f32_to_bf16(r1) << 16);
    o.y = (unsigned)f32_to_bf16(r2) | ((unsigned)f32_to_bf16(r3) << 16);
    *(uint2*)(ob + (size_t)posi * 256 + q) = o;
  }
}

// ---------------------------------------------------------------------------
// Prediction heads: direct conv3x3 + bias on NHWC bf16 input; optional
// exp(acc * scale). One thread per output element (B x Ko x H x W).
// ---------------------------------------------------------------------------
__global__ __launch_bounds__(256)
void head_conv_kernel(const u16* __restrict__ x,      // NHWC bf16
                      const float* __restrict__ wt,   // [Ko][256][3][3]
                      const float* __restrict__ bias,
                      float* __restrict__ out,
                      int Ko, int H, int W, int HW,
                      const float* __restrict__ scale_ptr, int do_exp) {
  const int idx = blockIdx.x * blockDim.x + threadIdx.x;
  const int total = 2 * Ko * HW;
  if (idx >= total) return;
  int t = idx;
  const int wpos = t % W; t /= W;
  const int hpos = t % H; t /= H;
  const int o    = t % Ko;
  const int b    = t / Ko;
  float acc = bias[o];
  const float* wo = wt + (size_t)o * 2304;
  #pragma unroll
  for (int kh = 0; kh < 3; ++kh) {
    const int hh = hpos + kh - 1;
    if (hh < 0 || hh >= H) continue;
    #pragma unroll
    for (int kw = 0; kw < 3; ++kw) {
      const int ww = wpos + kw - 1;
      if (ww < 0 || ww >= W) continue;
      const u16* xr = x + ((size_t)b * HW + (size_t)hh * W + ww) * 256;
      const float* wc = wo + kh * 3 + kw;
      for (int ci = 0; ci < 256; ++ci)
        acc += wc[ci * 9] * bf16_to_f32(xr[ci]);
    }
  }
  if (do_exp) acc = expf(acc * (*scale_ptr));
  out[idx] = acc;
}

// ---------------------------------------------------------------------------
extern "C" void kernel_launch(void* const* d_in, const int* in_sizes, int n_in,
                              void* d_out, int out_size, void* d_ws, size_t ws_size,
                              hipStream_t stream) {
  (void)in_sizes; (void)n_in; (void)out_size; (void)ws_size;
  const float* feats[5] = {(const float*)d_in[0], (const float*)d_in[1],
                           (const float*)d_in[2], (const float*)d_in[3],
                           (const float*)d_in[4]};
  const float* cls_tw_w   = (const float*)d_in[5];
  const float* cls_tw_g   = (const float*)d_in[6];
  const float* cls_tw_b   = (const float*)d_in[7];
  const float* box_tw_w   = (const float*)d_in[8];
  const float* box_tw_g   = (const float*)d_in[9];
  const float* box_tw_b   = (const float*)d_in[10];
  const float* cls_pred_w = (const float*)d_in[11];
  const float* cls_pred_b = (const float*)d_in[12];
  const float* box_pred_w = (const float*)d_in[13];
  const float* box_pred_b = (const float*)d_in[14];
  const float* ctr_pred_w = (const float*)d_in[15];
  const float* ctr_pred_b = (const float*)d_in[16];
  const float* scales     = (const float*)d_in[17];
  float* out = (float*)d_out;

  static const int Hs[5] = {100, 50, 25, 13, 7};

  // Workspace layout (all offsets 256B-aligned)
  const size_t WPACK_ELEMS = (size_t)4 * 9 * 8 * 16 * 32 * 16;   // per tower
  char* ws = (char*)d_ws;
  u16*   wpCls = (u16*)ws;
  u16*   wpBox = (u16*)(ws + WPACK_ELEMS * 2);
  float* buf0  = (float*)(ws + WPACK_ELEMS * 4);                  // NHWC f32
  u16*   buf1  = (u16*)(ws + WPACK_ELEMS * 4 + (size_t)2*256*100*100*4); // NHWC bf16
  u16*   bufF  = buf1 + (size_t)2 * 256 * 100 * 100;              // NHWC bf16 feat
  u16*   zpage = bufF + (size_t)2 * 256 * 100 * 100;              // 1KB zeros

  // Output offsets: logits[5], bboxes[5], centers[5] concatenated flat
  size_t logit_off[5], bbox_off[5], ctr_off[5];
  size_t acc = 0;
  for (int i = 0; i < 5; ++i) { logit_off[i] = acc; acc += (size_t)2 * 20 * Hs[i] * Hs[i]; }
  for (int i = 0; i < 5; ++i) { bbox_off[i]  = acc; acc += (size_t)2 * 4  * Hs[i] * Hs[i]; }
  for (int i = 0; i < 5; ++i) { ctr_off[i]   = acc; acc += (size_t)2 * 1  * Hs[i] * Hs[i]; }

  zero_page_kernel<<<1, 256, 0, stream>>>((unsigned*)zpage);

  const int repack_total = (int)WPACK_ELEMS;
  repack_weights_kernel<<<(repack_total + 255) / 256, 256, 0, stream>>>(
      cls_tw_w, wpCls, repack_total);
  repack_weights_kernel<<<(repack_total + 255) / 256, 256, 0, stream>>>(
      box_tw_w, wpBox, repack_total);

  const size_t LAYER_STRIDE = (size_t)9 * 8 * 16 * 32 * 16;

  for (int i = 0; i < 5; ++i) {
    const int H = Hs[i], W = Hs[i], HW = H * W;
    const int npos = 2 * HW;
    const int ntiles = (npos + 31) / 32;

    const int total8 = 2 * HW * 32;
    nchw_to_nhwc_bf16_kernel<<<(total8 + 255) / 256, 256, 0, stream>>>(
        feats[i], bufF, HW, total8);

    for (int tw = 0; tw < 2; ++tw) {
      const u16* wp    = tw ? wpBox : wpCls;
      const float* gms = tw ? box_tw_g : cls_tw_g;
      const float* bts = tw ? box_tw_b : cls_tw_b;
      const u16* cur = bufF;
      for (int l = 0; l < 4; ++l) {
        conv3x3_wmma_kernel<<<ntiles, 128, 0, stream>>>(
            cur, wp + (size_t)l * LAYER_STRIDE, buf0, zpage, H, W, HW, npos);
        gn_relu_kernel<<<64, 256, 0, stream>>>(
            buf0, gms + l * 256, bts + l * 256, buf1, HW);
        cur = buf1;
      }
      if (tw == 0) {
        int tot = 2 * 20 * HW;
        head_conv_kernel<<<(tot + 255) / 256, 256, 0, stream>>>(
            buf1, cls_pred_w, cls_pred_b, out + logit_off[i], 20, H, W, HW,
            (const float*)nullptr, 0);
        tot = 2 * 1 * HW;
        head_conv_kernel<<<(tot + 255) / 256, 256, 0, stream>>>(
            buf1, ctr_pred_w, ctr_pred_b, out + ctr_off[i], 1, H, W, HW,
            (const float*)nullptr, 0);
      } else {
        int tot = 2 * 4 * HW;
        head_conv_kernel<<<(tot + 255) / 256, 256, 0, stream>>>(
            buf1, box_pred_w, box_pred_b, out + bbox_off[i], 4, H, W, HW,
            scales + i, 1);
      }
    }
  }
}
